// LocalSpatioTemporalAttentionPatchKeras_58119497450440
// MI455X (gfx1250) — matrix operand order repaired and checked by reading, not compiled, not run on hardware
//
#include <hip/hip_runtime.h>
#include <hip/hip_fp16.h>

typedef __attribute__((ext_vector_type(16))) _Float16 v16h;
typedef __attribute__((ext_vector_type(8)))  _Float16 v8h;
typedef __attribute__((ext_vector_type(8)))  float    v8f;

#define TT 3
#define BB 2
#define HH 64
#define WW 64
#define CC 128
#define RYr 2
#define RXr 2
#define PWp 5
#define PPp 25
#define NKEYS (TT*PPp)   // 75
#define KDd 16
#define SA_LD (CC + 8)   // LDS row stride in halves: 272B -> 4-bank rotation/row

// ---------------------------------------------------------------------------
// prep: Xk = f16(x + temp_emb[t]), Xv = f16(x)
// ---------------------------------------------------------------------------
__global__ __launch_bounds__(256)
void prep_kv16(const float* __restrict__ X, const float* __restrict__ temp,
               _Float16* __restrict__ Xk, _Float16* __restrict__ Xv, int n)
{
    int i = blockIdx.x * blockDim.x + threadIdx.x;
    if (i >= n) return;
    int c = i & (CC - 1);
    int t = i / (BB * HH * WW * CC);
    float v = X[i];
    Xv[i] = (_Float16)v;
    Xk[i] = (_Float16)(v + temp[t * CC + c]);
}

// ---------------------------------------------------------------------------
// prep: Xq = f16(x[1] + temp_emb[1] + spatial_emb[sidx(y,x)])
// ---------------------------------------------------------------------------
__global__ __launch_bounds__(256)
void prep_q16(const float* __restrict__ X, const float* __restrict__ temp,
              const float* __restrict__ semb, _Float16* __restrict__ Xq, int n)
{
    int i = blockIdx.x * blockDim.x + threadIdx.x;
    if (i >= n) return;
    int c   = i & (CC - 1);
    int pix = i >> 7;                 // b*H*W + y*W + x
    int x   = pix % WW;
    int y   = (pix / WW) % HH;
    int yr  = y - (min(max(y, RYr), HH - 1 - RYr)) + RYr;
    int xr  = x - (min(max(x, RXr), WW - 1 - RXr)) + RXr;
    int p   = yr * PWp + xr;
    float v = X[(BB * HH * WW * CC) + i];     // frame q_idx = 1
    Xq[i] = (_Float16)(v + temp[CC + c] + semb[p * CC + c]);
}

// ---------------------------------------------------------------------------
// prep: transpose all 4 weight matrices to N-major f16 (BT[n][k])
// ---------------------------------------------------------------------------
__global__ __launch_bounds__(256)
void prep_wt(const float* __restrict__ Wq, const float* __restrict__ Wk,
             const float* __restrict__ Wv, const float* __restrict__ Wo,
             _Float16* __restrict__ WqT, _Float16* __restrict__ WkT,
             _Float16* __restrict__ WvT, _Float16* __restrict__ WoT)
{
    int i = blockIdx.x * blockDim.x + threadIdx.x;
    if (i >= CC * CC) return;
    int k = i & (CC - 1);           // contraction index within BT row
    WqT[i] = (_Float16)Wq[k * CC + (i >> 7)];
    WkT[i] = (_Float16)Wk[k * CC + (i >> 7)];
    WvT[i] = (_Float16)Wv[k * CC + (i >> 7)];
    WoT[i] = (_Float16)Wo[k * CC + (i >> 7)];
}

// ---------------------------------------------------------------------------
// prep: Ksp[p][hd] = spatial_emb[p]·Wk  (tiny, f32 VALU)
// ---------------------------------------------------------------------------
__global__ __launch_bounds__(256)
void prep_ksp(const float* __restrict__ semb, const float* __restrict__ Wk,
              float* __restrict__ Ksp)
{
    int i = blockIdx.x * blockDim.x + threadIdx.x;
    if (i >= PPp * CC) return;
    int p = i >> 7, hd = i & (CC - 1);
    float s = 0.f;
    #pragma unroll 8
    for (int c = 0; c < CC; ++c) s += semb[p * CC + c] * Wk[c * CC + hd];
    Ksp[i] = s;
}

// ---------------------------------------------------------------------------
// WMMA GEMM: C[M x 128] = A16[M x 128] * B (given as BT16[128 x 128], N-major)
//            + bias[128].
// A tile (16x128, 4KB) staged once per block via ASYNC global->LDS copy
// (256 lanes x b128), padded stride to avoid LDS bank conflicts; all 8 waves
// then read fragments with ds_load. B read direct (L0-resident, 32KB total).
// ---------------------------------------------------------------------------
template<bool OUT_F16>
__global__ __launch_bounds__(256)
void gemm_wmma(const _Float16* __restrict__ A, const _Float16* __restrict__ BT,
               const float* __restrict__ bias,
               float* __restrict__ Cf, _Float16* __restrict__ Ch, int M)
{
    __shared__ _Float16 sA[16 * SA_LD];            // 4352 B

    const int tid  = threadIdx.x;
    const int lane = tid & 31;
    const int wv   = tid >> 5;             // ntile 0..7
    const int mt   = blockIdx.x;           // mtile
    const int r16  = lane & 15;            // A row / B col within tile
    const int kh   = lane >> 4;            // K-half select

    // --- async stage A tile: thread -> (row = tid/16, 16B chunk = tid%16) ---
    {
        const int arow  = tid >> 4;                 // 0..15
        const int achk  = tid & 15;                 // 0..15 (16B chunks)
        unsigned long long ga =
            (unsigned long long)(A + (mt * 16 + arow) * CC + achk * 8);
        unsigned ldsA = (unsigned)(size_t)(&sA[0]) +
                        (unsigned)(arow * (SA_LD * 2) + achk * 16);
        asm volatile("global_load_async_to_lds_b128 %0, %1, off"
                     :: "v"(ldsA), "v"(ga) : "memory");
    }
    asm volatile("s_wait_asynccnt 0" ::: "memory");
    __syncthreads();

    const _Float16* arow = &sA[r16 * SA_LD];
    const _Float16* __restrict__ bcol = BT + (wv * 16 + r16) * CC;

    __builtin_prefetch(A + (mt + 1) * 16 * CC, 0, 0);   // speculative: next M tile

    v8f c = {};
    #pragma unroll
    for (int kb = 0; kb < CC; kb += 32) {
        v16h a, b;
        // A 16x32 f16 fragment: lane<16 holds K {0..7,16..23}+kb, lane>=16 the other half
        ((v8h*)&a)[0] = *(const v8h*)(arow + kb + kh * 8);
        ((v8h*)&a)[1] = *(const v8h*)(arow + kb + 16 + kh * 8);
        // B 32x16 f16 fragment: lane holds col r16, K {0..7,16..23}+kb (lane<16) or the other half (N-major BT)
        ((v8h*)&b)[0] = *(const v8h*)(bcol + kb + kh * 8);
        ((v8h*)&b)[1] = *(const v8h*)(bcol + kb + 16 + kh * 8);
        c = __builtin_amdgcn_wmma_f32_16x16x32_f16(
                false, a, false, b, (short)0, c, false, false);
    }

    const int n  = wv * 16 + r16;
    const float bn = bias[n];
    #pragma unroll
    for (int v = 0; v < 8; ++v) {
        int m = mt * 16 + v + 8 * kh;      // C/D layout: M = v + 8*(lane/16)
        float val = c[v] + bn;
        if (OUT_F16) Ch[m * CC + n] = (_Float16)val;
        else         Cf[m * CC + n] = val;
    }
}

// ---------------------------------------------------------------------------
// attention: one block (128 thr = 8 heads x 16 dims) per query pixel,
// online softmax over 75 keys, 16-lane shfl_xor dot reduction (wave32-native)
// ---------------------------------------------------------------------------
__global__ __launch_bounds__(128)
void attn_kernel(const _Float16* __restrict__ Q16,
                 const _Float16* __restrict__ Kb16,
                 const _Float16* __restrict__ Vb16,
                 const float*    __restrict__ Ksp,
                 _Float16*       __restrict__ ctx16)
{
    const int pix = blockIdx.x;                  // b*H*W + y*W + x
    const int b   = pix / (HH * WW);
    const int rem = pix % (HH * WW);
    const int y = rem / WW, x = rem % WW;
    const int hd  = threadIdx.x;                 // head*16 + d

    const float q  = (float)Q16[pix * CC + hd];
    const int   y0 = min(max(y, RYr), HH - 1 - RYr) - RYr;
    const int   x0 = min(max(x, RXr), WW - 1 - RXr) - RXr;

    float m = -1e30f, l = 0.f, acc = 0.f;
    for (int n = 0; n < NKEYS; ++n) {
        int t  = n / PPp, p = n % PPp;
        int ky = y0 + p / PWp;
        int kx = x0 + p % PWp;
        int kidx = (((t * BB + b) * HH + ky) * WW + kx) * CC + hd;
        float kv = (float)Kb16[kidx] + Ksp[p * CC + hd];
        float s  = q * kv;
        #pragma unroll
        for (int off = 8; off > 0; off >>= 1) s += __shfl_xor(s, off, 16);
        s *= 0.25f;                              // 1/sqrt(KD=16)
        float vv = (float)Vb16[kidx];
        float mn = fmaxf(m, s);
        float sc = __expf(m - mn);
        float w  = __expf(s - mn);
        l   = l * sc + w;
        acc = acc * sc + w * vv;
        m   = mn;
    }
    ctx16[pix * CC + hd] = (_Float16)(acc / l);
}

// ---------------------------------------------------------------------------
extern "C" void kernel_launch(void* const* d_in, const int* in_sizes, int n_in,
                              void* d_out, int out_size, void* d_ws, size_t ws_size,
                              hipStream_t stream) {
    const float* all_values = (const float*)d_in[0];
    const float* temp_emb   = (const float*)d_in[1];
    const float* spat_emb   = (const float*)d_in[2];
    const float* Wq = (const float*)d_in[3];
    const float* bq = (const float*)d_in[4];
    const float* Wk = (const float*)d_in[5];
    const float* bk = (const float*)d_in[6];
    const float* Wv = (const float*)d_in[7];
    const float* bv = (const float*)d_in[8];
    const float* Wo = (const float*)d_in[9];
    const float* bo = (const float*)d_in[10];
    float* out = (float*)d_out;

    const int BHW  = BB * HH * WW;               // 8192
    const int MKV  = TT * BHW;                   // 24576
    const int nKV  = MKV * CC;                   // 3145728
    const int nQ   = BHW * CC;                   // 1048576

    // workspace layout (all offsets multiples of 256 bytes)
    char* w = (char*)d_ws;
    size_t off = 0;
    _Float16* Xk16  = (_Float16*)(w + off); off += (size_t)nKV * 2;         // 6.29 MB
    _Float16* Xv16  = (_Float16*)(w + off); off += (size_t)nKV * 2;         // 6.29 MB
    _Float16* Xq16  = (_Float16*)(w + off); off += (size_t)nQ * 2;          // 2.10 MB
    _Float16* WqT   = (_Float16*)(w + off); off += CC * CC * 2;
    _Float16* WkT   = (_Float16*)(w + off); off += CC * CC * 2;
    _Float16* WvT   = (_Float16*)(w + off); off += CC * CC * 2;
    _Float16* WoT   = (_Float16*)(w + off); off += CC * CC * 2;
    float*    Ksp   = (float*)   (w + off); off += PPp * CC * 4;            // 12.8 KB
    _Float16* Kb16  = (_Float16*)(w + off); off += (size_t)nKV * 2;         // 6.29 MB
    _Float16* Vb16  = (_Float16*)(w + off); off += (size_t)nKV * 2;         // 6.29 MB
    _Float16* Q16   = (_Float16*)(w + off); off += (size_t)nQ * 2;          // 2.10 MB
    _Float16* ctx16 = (_Float16*)(w + off); off += (size_t)nQ * 2;          // 2.10 MB

    // 1) build f16 GEMM inputs (fold temp/spatial embeddings — linearity)
    prep_kv16<<<(nKV + 255) / 256, 256, 0, stream>>>(all_values, temp_emb, Xk16, Xv16, nKV);
    prep_q16 <<<(nQ  + 255) / 256, 256, 0, stream>>>(all_values, temp_emb, spat_emb, Xq16, nQ);
    prep_wt  <<<(CC * CC + 255) / 256, 256, 0, stream>>>(Wq, Wk, Wv, Wo, WqT, WkT, WvT, WoT);
    prep_ksp <<<(PPp * CC + 255) / 256, 256, 0, stream>>>(spat_emb, Wk, Ksp);

    // 2) WMMA projections: Kbase, Vbase (M=24576), Q (M=8192) -> f16
    gemm_wmma<true><<<MKV / 16, 256, 0, stream>>>(Xk16, WkT, bk, nullptr, Kb16, MKV);
    gemm_wmma<true><<<MKV / 16, 256, 0, stream>>>(Xv16, WvT, bv, nullptr, Vb16, MKV);
    gemm_wmma<true><<<BHW / 16, 256, 0, stream>>>(Xq16, WqT, bq, nullptr, Q16,  BHW);

    // 3) per-pixel online-softmax attention over 75 keys
    attn_kernel<<<BHW, 128, 0, stream>>>(Q16, Kb16, Vb16, Ksp, ctx16);

    // 4) WMMA output projection -> f32 d_out
    gemm_wmma<false><<<BHW / 16, 256, 0, stream>>>(ctx16, WoT, bo, out, nullptr, BHW);

    (void)in_sizes; (void)n_in; (void)out_size; (void)ws_size;
}
